// SupConLoss_83786222010855
// MI455X (gfx1250) — matrix operand order repaired and checked by reading, not compiled, not run on hardware
//
#include <hip/hip_runtime.h>
#include <hip/hip_bf16.h>

// ---------------------------------------------------------------------------
// SupCon loss, fused, for MI455X (gfx1250, wave32, WMMA).
//
// features: f32 [16, 1024, 2, 128] unit-norm; labels: i32 [16, 1024].
// Per batch: C = contrast [2048,128]; logits = C C^T / 0.07;
//   row contribution = (sum_{match,m!=n} logits)/cnt - log(sum_{m!=n} exp(logits))
//   loss = -(1/(16*2048)) * sum of contributions.
// Unit-norm rows => |logits| <= 14.29, so direct f32 exp is safe (the
// reference's row-max subtraction cancels algebraically in log_prob).
//
// Round-2 change: ping-pong double buffering of the B fragments + label so
// the 8x global_load_b128 prefetch for tile nt+1 overlaps the 4 WMMAs and
// exp epilogue of tile nt (partial s_wait_loadcnt instead of wait-0 per load).
// ---------------------------------------------------------------------------

typedef _Float16 v16h __attribute__((ext_vector_type(16)));
typedef _Float16 v8h  __attribute__((ext_vector_type(8)));
typedef _Float16 v4h  __attribute__((ext_vector_type(4)));
typedef float    v8f  __attribute__((ext_vector_type(8)));

#define BATCHES 16
#define BSZ     1024
#define NVIEW   2
#define NROWS   2048     // NVIEW * BSZ
#define DDIM    128
#define NT      (NROWS / 16)          // 128 column tiles
#define INV_T   14.285714285714286f   // 1 / 0.07

union AF { v16h v; v8h h[2]; };

// ---- kernel 1: f32 [16,1024,2,128] -> f16 contrast-ordered [16,2048,128] ---
// contrast row n of batch b = feat[b, n % 1024, n / 1024, :]; 4 elems/thread
__global__ void supcon_convert(const float* __restrict__ feat,
                               _Float16* __restrict__ c16) {
    int idx = blockIdx.x * blockDim.x + threadIdx.x;    // 0 .. 16*2048*32-1
    int k4 = idx & (DDIM / 4 - 1);
    int n  = (idx >> 5) & (NROWS - 1);
    int b  = idx >> 16;
    int s  = n & (BSZ - 1);
    int v  = n >> 10;
    const float4 x = *(const float4*)
        (feat + (((size_t)b * BSZ + s) * NVIEW + v) * DDIM + k4 * 4);
    v4h h = { (_Float16)x.x, (_Float16)x.y, (_Float16)x.z, (_Float16)x.w };
    *(v4h*)(c16 + (size_t)idx * 4) = h;
}

// ---- kernel 2: fused GEMM + masked-softmax-stats epilogue ------------------
__global__ __launch_bounds__(256)
void supcon_main(const _Float16* __restrict__ c16,
                 const int* __restrict__ labels,
                 float* __restrict__ partials) {
    const int b    = blockIdx.x >> 4;       // batch 0..15
    const int mt   = blockIdx.x & 15;       // row-tile of 128 rows
    const int wave = threadIdx.x >> 5;      // 0..7 (wave32)
    const int lane = threadIdx.x & 31;
    const int hi   = lane >> 4;             // lane half
    const int l16  = lane & 15;

    const _Float16* C   = c16 + (size_t)b * NROWS * DDIM;
    const int*      lab = labels + b * BSZ;
    const int       wrow = mt * 128 + wave * 16;   // this wave's 16-row strip

    // --- preload A fragments for all 4 K-chunks (ISA 16-bit A layout) ------
    // element e<8  : C[m][kc*32 + 8*hi  + e]
    // element e>=8 : C[m][kc*32 + 16 + 8*hi + (e-8)]
    AF a[4];
    {
        const _Float16* arow = C + (size_t)(wrow + l16) * DDIM;
#pragma unroll
        for (int kc = 0; kc < 4; ++kc) {
            a[kc].h[0] = *(const v8h*)(arow + kc * 32 + hi * 8);
            a[kc].h[1] = *(const v8h*)(arow + kc * 32 + 16 + hi * 8);
        }
    }

    // C-matrix layout: VGPR r, lane half hi -> row (wrow + r + 8*hi), col = l16
    int myrow[8], rlab[8];
#pragma unroll
    for (int r = 0; r < 8; ++r) {
        myrow[r] = wrow + r + 8 * hi;
        rlab[r]  = lab[myrow[r] & (BSZ - 1)];
    }

    float esum[8], msum[8], cnt[8];
#pragma unroll
    for (int r = 0; r < 8; ++r) { esum[r] = 0.f; msum[r] = 0.f; cnt[r] = 0.f; }

    // B-fragment loader (same k order as A: e<8 -> kc*32+8*hi+e, e>=8 -> kc*32+16+8*hi+(e-8))
    auto loadB = [&](int nt, AF* bf, int& clab) {
        const _Float16* brow = C + (size_t)(nt * 16 + l16) * DDIM;
#pragma unroll
        for (int kc = 0; kc < 4; ++kc) {
            bf[kc].h[0] = *(const v8h*)(brow + kc * 32 + hi * 8);
            bf[kc].h[1] = *(const v8h*)(brow + kc * 32 + 16 + hi * 8);
        }
        clab = lab[(nt * 16 + l16) & (BSZ - 1)];
    };

    // 4 chained WMMAs + per-lane stats epilogue for one 16-column tile
    auto tile = [&](const AF* bf, int nt, int clab) {
        v8f acc = {};
#pragma unroll
        for (int kc = 0; kc < 4; ++kc)
            acc = __builtin_amdgcn_wmma_f32_16x16x32_f16(
                false, a[kc].v, false, bf[kc].v, (short)0, acc, false, false);
        const int col = nt * 16 + l16;
#pragma unroll
        for (int r = 0; r < 8; ++r) {
            float s     = acc[r] * INV_T;
            bool  diag  = (myrow[r] == col);
            float e     = __expf(s);
            esum[r]    += diag ? 0.f : e;
            bool  match = (rlab[r] == clab) && !diag;
            msum[r]    += match ? s : 0.f;
            cnt[r]     += match ? 1.f : 0.f;
        }
    };

    // --- software-pipelined sweep over 128 column tiles (ping-pong B) -----
    AF  b0[4], b1[4];
    int clab0, clab1;
    loadB(0, b0, clab0);
    for (int nt = 0; nt < NT; nt += 2) {
        loadB(nt + 1, b1, clab1);              // prefetch odd tile
        tile(b0, nt, clab0);                   // compute even tile
        if (nt + 2 < NT) loadB(nt + 2, b0, clab0);   // prefetch next even tile
        tile(b1, nt + 1, clab1);               // compute odd tile
    }

    // --- one cross-lane reduction over the 16 lanes of each half ----------
#pragma unroll
    for (int r = 0; r < 8; ++r) {
#pragma unroll
        for (int m = 1; m < 16; m <<= 1) {
            esum[r] += __shfl_xor(esum[r], m, 32);
            msum[r] += __shfl_xor(msum[r], m, 32);
            cnt[r]  += __shfl_xor(cnt[r],  m, 32);
        }
    }

    // lanes 0 and 16 now hold full row stats for rows r and r+8
    float wsum = 0.f;
    if (l16 == 0) {
#pragma unroll
        for (int r = 0; r < 8; ++r)
            wsum += msum[r] / cnt[r] - __logf(esum[r]);   // cnt >= 1 (other view)
    }
#pragma unroll
    for (int m = 1; m < 32; m <<= 1) wsum += __shfl_xor(wsum, m, 32);

    __shared__ float wpart[8];
    if (lane == 0) wpart[wave] = wsum;
    __syncthreads();
    if (threadIdx.x == 0) {
        float t = 0.f;
        for (int w = 0; w < 8; ++w) t += wpart[w];        // fixed order: deterministic
        partials[blockIdx.x] = t;
    }
}

// ---- kernel 3: deterministic fixed-order final reduction -------------------
__global__ void supcon_finalize(const float* __restrict__ partials,
                                float* __restrict__ out) {
    if (blockIdx.x == 0 && threadIdx.x == 0) {
        float t = 0.f;
        for (int i = 0; i < 256; ++i) t += partials[i];
        out[0] = -t / (float)(BATCHES * NROWS);
    }
}

extern "C" void kernel_launch(void* const* d_in, const int* in_sizes, int n_in,
                              void* d_out, int out_size, void* d_ws, size_t ws_size,
                              hipStream_t stream) {
    const float* feat   = (const float*)d_in[0];   // [16,1024,2,128] f32
    const int*   labels = (const int*)d_in[1];     // [16,1024] i32

    _Float16* c16      = (_Float16*)d_ws;                       // 8 MB f16 contrast
    float*    partials = (float*)((char*)d_ws +
                          (size_t)BATCHES * NROWS * DDIM * sizeof(_Float16));

    const int convThreads = BATCHES * NROWS * (DDIM / 4);       // 1,048,576
    supcon_convert<<<convThreads / 256, 256, 0, stream>>>(feat, c16);
    supcon_main<<<256, 256, 0, stream>>>(c16, labels, partials);
    supcon_finalize<<<1, 32, 0, stream>>>(partials, (float*)d_out);
}
